// CrossAttentionBlock_987842478101
// MI455X (gfx1250) — matrix operand order repaired and checked by reading, not compiled, not run on hardware
//
#include <hip/hip_runtime.h>
#include <hip/hip_bf16.h>

// ---------------------------------------------------------------------------
// CDNA5 (gfx1250) fused cross-attention:
//   Kernel 1: f32 GEMM (X @ W^T) -> f16 in d_ws; Q/K head-major [b][h][l][64],
//             V transposed [b][h][dh][1024] so AV B-fragments are contiguous.
//   Kernel 2: per-(b,h) flash attention, 64-key tiles, two key sources,
//             online softmax, async global->LDS staging (ASYNCcnt path),
//             all matmuls via v_wmma_f32_16x16x32_f16 (wave32).
// ---------------------------------------------------------------------------

typedef __attribute__((ext_vector_type(16))) _Float16 v16h;
typedef __attribute__((ext_vector_type(8)))  float    v8f;
typedef __attribute__((__vector_size__(4 * sizeof(int)))) int v4i_b128;

#define HIDDEN   512
#define HEADS    8
#define HEADDIM  64
#define SEQ      1024
#define BATCH    4
#define NEG_INF_ADD (-10000.0f)

#if defined(__has_builtin)
#if __has_builtin(__builtin_amdgcn_global_load_async_to_lds_b128)
#define HAVE_ASYNC_LDS 1
#endif
#endif

static __device__ inline v8f wmma16x16x32(v16h a, v16h b, v8f c) {
  // (neg_a, A, neg_b, B, c_mod, C, reuse_a, reuse_b)
  return __builtin_amdgcn_wmma_f32_16x16x32_f16(false, a, false, b,
                                                (short)0, c, false, false);
}

#ifdef HAVE_ASYNC_LDS
typedef __attribute__((address_space(1))) v4i_b128* g_b128_ptr;
typedef __attribute__((address_space(3))) v4i_b128* l_b128_ptr;
static __device__ inline void async_b128(const _Float16* g, _Float16* l) {
  __builtin_amdgcn_global_load_async_to_lds_b128(
      (g_b128_ptr)g, (l_b128_ptr)l, 0, 0);
}
static __device__ inline void async_wait0() {
#if __has_builtin(__builtin_amdgcn_s_wait_asynccnt)
  __builtin_amdgcn_s_wait_asynccnt(0);
#else
  asm volatile("s_wait_asynccnt 0x0" ::: "memory");
#endif
}
#endif

// ---------------------------------------------------------------------------
// Projection GEMM:  C[m][n] = sum_k X[m][k] * W[n][k]  (m=(b,l), n=(h,dh))
// transposedOut==0: out[((b*8+h)*1024+l)*64+dh]   (Q, K)
// transposedOut==1: out[((b*8+h)*64+dh)*1024+l]   (V^T)
// Block: 256 threads (8 waves). Block tile 128(M) x 128(N); wave tile 32x64.
// ---------------------------------------------------------------------------
__global__ __launch_bounds__(256)
void proj_gemm_kernel(const float* __restrict__ X,
                      const float* __restrict__ W,
                      _Float16* __restrict__ out,
                      int transposedOut) {
  __shared__ _Float16 Xs[128 * 48];   // 32-wide K tile, stride 48 halves (96B)
  __shared__ _Float16 Ws[128 * 48];

  const int tid  = threadIdx.x;
  const int lane = tid & 31;
  const int wv   = tid >> 5;
  const int mBase = blockIdx.x * 128;
  const int nBase = blockIdx.y * 128;
  const int waveM = (wv >> 1) * 32;   // 4 waves along M
  const int waveN = (wv & 1) * 64;    // 2 waves along N

  const int lc    = lane & 15;
  const int half  = lane >> 4;
  const int kOffA = half * 8;
  const int kOffB = half * 16;

  v8f acc[2][4];
#pragma unroll
  for (int md = 0; md < 2; ++md)
#pragma unroll
    for (int nd = 0; nd < 4; ++nd)
#pragma unroll
      for (int r = 0; r < 8; ++r) acc[md][nd][r] = 0.0f;

  for (int k0 = 0; k0 < HIDDEN; k0 += 32) {
#pragma unroll
    for (int i = 0; i < 16; ++i) {
      int idx = i * 256 + tid;          // 0..4095
      int row = idx >> 5;
      int col = idx & 31;
      Xs[row * 48 + col] = (_Float16)X[(size_t)(mBase + row) * HIDDEN + k0 + col];
      Ws[row * 48 + col] = (_Float16)W[(size_t)(nBase + row) * HIDDEN + k0 + col];
    }
    __syncthreads();

    v16h afr[2];
#pragma unroll
    for (int md = 0; md < 2; ++md) {
      int base = (waveM + md * 16 + lc) * 48;
#pragma unroll
      for (int j = 0; j < 8; ++j) {
        afr[md][j]     = Xs[base + kOffA + j];
        afr[md][8 + j] = Xs[base + kOffA + 16 + j];
      }
    }
    v16h bfr[4];
#pragma unroll
    for (int nd = 0; nd < 4; ++nd) {
      int base = (waveN + nd * 16 + lc) * 48;
#pragma unroll
      for (int j = 0; j < 16; ++j) bfr[nd][j] = Ws[base + kOffA + (j & 7) + ((j >> 3) << 4)];
    }
#pragma unroll
    for (int md = 0; md < 2; ++md)
#pragma unroll
      for (int nd = 0; nd < 4; ++nd)
        acc[md][nd] = wmma16x16x32(afr[md], bfr[nd], acc[md][nd]);
    __syncthreads();
  }

#pragma unroll
  for (int md = 0; md < 2; ++md) {
#pragma unroll
    for (int nd = 0; nd < 4; ++nd) {
#pragma unroll
      for (int r = 0; r < 8; ++r) {
        int m = mBase + waveM + md * 16 + r + 8 * half;   // (b,l)
        int n = nBase + waveN + nd * 16 + lc;             // (h,dh)
        int b = m >> 10, l = m & 1023;
        int h = n >> 6,  dh = n & 63;
        size_t idx = transposedOut
            ? (((size_t)(b * HEADS + h)) * HEADDIM + dh) * SEQ + l
            : (((size_t)(b * HEADS + h)) * SEQ + l) * HEADDIM + dh;
        out[idx] = (_Float16)acc[md][nd][r];
      }
    }
  }
}

// ---------------------------------------------------------------------------
// Flash attention for one output side, 64-key tiles:
//   out[b][l][h*64+dh] = rowmask * 0.5 * ( softmax(Q K1^T + mask)·V1
//                                        + softmax(Q K2^T + mask)·V2 )
// Grid: 256 blocks = (b,h,qblock of 128). Block: 256 threads = 8 waves,
// each wave owns 16 q rows.  K tile [64 keys][64 dh]; V^T tile [64 dh][64 k].
// ---------------------------------------------------------------------------
__global__ __launch_bounds__(256)
void flash_attn_kernel(const _Float16* __restrict__ Q,    // [bh][l][64]
                       const _Float16* __restrict__ K1,   // [bh][l][64]
                       const _Float16* __restrict__ V1t,  // [bh][dh][1024]
                       const _Float16* __restrict__ K2,
                       const _Float16* __restrict__ V2t,
                       const float* __restrict__ maskRow,
                       const float* __restrict__ maskC1,
                       const float* __restrict__ maskC2,
                       float* __restrict__ out) {
  __shared__ _Float16 Ks[64 * HEADDIM];    // 8 KB  [key][dh]
  __shared__ _Float16 Vts[HEADDIM * 64];   // 8 KB  [dh][key]
  __shared__ _Float16 Ps[8][16 * 64];      // 16 KB per-wave P bounce

  const int tid  = threadIdx.x;
  const int lane = tid & 31;
  const int wv   = tid >> 5;

  const int blk = blockIdx.x;
  const int qb  = blk & 7;
  const int bh  = blk >> 3;          // b*8 + h
  const int h   = bh & 7;
  const int b   = bh >> 3;
  const int qBase = qb * 128 + wv * 16;

  const size_t bhOff = (size_t)bh * SEQ * HEADDIM;   // 65536, same for V^T
  const int lc    = lane & 15;
  const int half  = lane >> 4;
  const int kOffA = half * 8;
  const int kOffB = half * 16;

  // Resident Q A-fragments: 16 rows x 64 dh (two K=32 chunks).
  v16h aq[2];
  {
    const _Float16* qp = Q + bhOff + (size_t)(qBase + lc) * HEADDIM;
#pragma unroll
    for (int c = 0; c < 2; ++c)
#pragma unroll
      for (int j = 0; j < 8; ++j) {
        aq[c][j]     = qp[c * 32 + kOffA + j];
        aq[c][8 + j] = qp[c * 32 + kOffA + 16 + j];
      }
  }

  v8f outAcc[4];
#pragma unroll
  for (int nd = 0; nd < 4; ++nd)
#pragma unroll
    for (int r = 0; r < 8; ++r) outAcc[nd][r] = 0.0f;

  for (int s = 0; s < 2; ++s) {
    const _Float16* Kt = s ? K2 : K1;
    const _Float16* Vt = s ? V2t : V1t;
    const float*    mc = s ? maskC2 : maskC1;

    float mrun[8], lrun[8];
#pragma unroll
    for (int r = 0; r < 8; ++r) { mrun[r] = -3.0e38f; lrun[r] = 0.0f; }
    v8f oacc[4];
#pragma unroll
    for (int nd = 0; nd < 4; ++nd)
#pragma unroll
      for (int r = 0; r < 8; ++r) oacc[nd][r] = 0.0f;

    for (int kt = 0; kt < SEQ / 64; ++kt) {
      const int keyBase = kt * 64;

      __syncthreads();   // previous tile fully consumed
      {
        // Stage K [64][64] and V^T [64][64] : 4096 halves each,
        // 512 b128 chunks each, 2 per thread per tensor.
        const _Float16* kg = Kt + bhOff + (size_t)keyBase * HEADDIM;
#pragma unroll
        for (int i = 0; i < 2; ++i) {
          int idx = i * 256 + tid;           // chunk id 0..511 (8 halves each)
          int dh  = idx >> 3;                // V^T row
          int c   = idx & 7;
          const _Float16* gv = Vt + bhOff + (size_t)dh * SEQ + keyBase + c * 8;
#ifdef HAVE_ASYNC_LDS
          async_b128(kg + idx * 8, &Ks[idx * 8]);
          async_b128(gv,           &Vts[idx * 8]);   // dh*64 + c*8 == idx*8
#else
#pragma unroll
          for (int j = 0; j < 8; ++j) {
            Ks[idx * 8 + j]  = kg[idx * 8 + j];
            Vts[idx * 8 + j] = gv[j];
          }
#endif
        }
#ifdef HAVE_ASYNC_LDS
        async_wait0();
#endif
      }
      __syncthreads();

      // Additive key masks for the four 16-key sub-tiles.
      float madd[4];
#pragma unroll
      for (int sub = 0; sub < 4; ++sub)
        madd[sub] = (mc[b * SEQ + keyBase + sub * 16 + lc] > 0.0f)
                        ? 0.0f : NEG_INF_ADD;

      // S = Q K^T : four 16x16 key sub-tiles, K-dim 64 = 2 chunks of 32.
      v8f sfr[4];
#pragma unroll
      for (int sub = 0; sub < 4; ++sub)
#pragma unroll
        for (int r = 0; r < 8; ++r) sfr[sub][r] = 0.0f;
#pragma unroll
      for (int c = 0; c < 2; ++c) {
#pragma unroll
        for (int sub = 0; sub < 4; ++sub) {
          v16h bk;
          int base = (sub * 16 + lc) * HEADDIM + c * 32 + kOffA;
#pragma unroll
          for (int j = 0; j < 16; ++j) bk[j] = Ks[base + (j & 7) + ((j >> 3) << 4)];
          sfr[sub] = wmma16x16x32(aq[c], bk, sfr[sub]);
        }
      }
#pragma unroll
      for (int sub = 0; sub < 4; ++sub)
#pragma unroll
        for (int r = 0; r < 8; ++r) sfr[sub][r] += madd[sub];

      // Online softmax update (row reduction across the 16-lane col group).
      float newm[8];
#pragma unroll
      for (int r = 0; r < 8; ++r) {
        float v = fmaxf(fmaxf(sfr[0][r], sfr[1][r]),
                        fmaxf(sfr[2][r], sfr[3][r]));
#pragma unroll
        for (int d = 1; d < 16; d <<= 1) v = fmaxf(v, __shfl_xor(v, d, 32));
        newm[r] = fmaxf(mrun[r], v);
      }
#pragma unroll
      for (int r = 0; r < 8; ++r) {
        float scale = __expf(mrun[r] - newm[r]);
        mrun[r] = newm[r];
        float rs = 0.0f;
#pragma unroll
        for (int sub = 0; sub < 4; ++sub) {
          float p = __expf(sfr[sub][r] - newm[r]);
          sfr[sub][r] = p;
          rs += p;
        }
#pragma unroll
        for (int d = 1; d < 16; d <<= 1) rs += __shfl_xor(rs, d, 32);
        lrun[r] = lrun[r] * scale + rs;
#pragma unroll
        for (int nd = 0; nd < 4; ++nd) oacc[nd][r] *= scale;
      }

      // Bounce P (C-layout) through per-wave LDS to build A-fragments.
      _Float16* pw = &Ps[wv][0];
#pragma unroll
      for (int r = 0; r < 8; ++r) {
        int row = r + 8 * half;
#pragma unroll
        for (int sub = 0; sub < 4; ++sub)
          pw[row * 64 + sub * 16 + lc] = (_Float16)sfr[sub][r];
      }
      v16h pa[2];
#pragma unroll
      for (int c2 = 0; c2 < 2; ++c2) {
        int base = lc * 64 + c2 * 32;
#pragma unroll
        for (int j = 0; j < 8; ++j) {
          pa[c2][j]     = pw[base + kOffA + j];
          pa[c2][8 + j] = pw[base + kOffA + 16 + j];
        }
      }

      // O += P · V : K-dim 64 keys (2 chunks), N = 64 dh (4 sub-tiles).
      // V^T tile rows are contiguous in keys -> b128 fragment loads.
#pragma unroll
      for (int c2 = 0; c2 < 2; ++c2) {
#pragma unroll
        for (int nd = 0; nd < 4; ++nd) {
          v16h vb;
          int base = (nd * 16 + lc) * 64 + c2 * 32 + kOffA;
#pragma unroll
          for (int j = 0; j < 16; ++j) vb[j] = Vts[base + (j & 7) + ((j >> 3) << 4)];
          oacc[nd] = wmma16x16x32(pa[c2], vb, oacc[nd]);
        }
      }
    }

    // Fold this source: outAcc += 0.5 * O / l
#pragma unroll
    for (int r = 0; r < 8; ++r) {
      float inv = 0.5f / fmaxf(lrun[r], 1e-30f);
#pragma unroll
      for (int nd = 0; nd < 4; ++nd) outAcc[nd][r] += oacc[nd][r] * inv;
    }
  }

  // Store with query-row mask.
#pragma unroll
  for (int r = 0; r < 8; ++r) {
    int qg = qBase + r + 8 * half;
    float rm = (maskRow[b * SEQ + qg] > 0.0f) ? 1.0f : 0.0f;
#pragma unroll
    for (int nd = 0; nd < 4; ++nd) {
      int dh = nd * 16 + lc;
      out[((size_t)(b * SEQ + qg)) * HIDDEN + h * HEADDIM + dh] =
          outAcc[nd][r] * rm;
    }
  }
}

// ---------------------------------------------------------------------------
// Launch
// ---------------------------------------------------------------------------
extern "C" void kernel_launch(void* const* d_in, const int* in_sizes, int n_in,
                              void* d_out, int out_size, void* d_ws, size_t ws_size,
                              hipStream_t stream) {
  const float* input1 = (const float*)d_in[0];
  const float* mask1  = (const float*)d_in[1];
  const float* input2 = (const float*)d_in[2];
  const float* mask2  = (const float*)d_in[3];
  const float* Wq1    = (const float*)d_in[4];
  const float* Wk1    = (const float*)d_in[5];
  const float* Wv1    = (const float*)d_in[6];
  const float* Wq2    = (const float*)d_in[7];
  const float* Wk2    = (const float*)d_in[8];
  const float* Wv2    = (const float*)d_in[9];
  float* out = (float*)d_out;

  const size_t T = (size_t)BATCH * HEADS * SEQ * HEADDIM;  // 2,097,152 f16 each
  _Float16* ws = (_Float16*)d_ws;
  _Float16* q1  = ws + 0 * T;
  _Float16* k1  = ws + 1 * T;
  _Float16* v1t = ws + 2 * T;
  _Float16* q2  = ws + 3 * T;
  _Float16* k2  = ws + 4 * T;
  _Float16* v2t = ws + 5 * T;

  dim3 gGemm(32, 4, 1);   // 4096/128 M-tiles, 512/128 N-tiles
  dim3 blk(256, 1, 1);
  proj_gemm_kernel<<<gGemm, blk, 0, stream>>>(input1, Wq1, q1,  0);
  proj_gemm_kernel<<<gGemm, blk, 0, stream>>>(input1, Wk1, k1,  0);
  proj_gemm_kernel<<<gGemm, blk, 0, stream>>>(input1, Wv1, v1t, 1);
  proj_gemm_kernel<<<gGemm, blk, 0, stream>>>(input2, Wq2, q2,  0);
  proj_gemm_kernel<<<gGemm, blk, 0, stream>>>(input2, Wk2, k2,  0);
  proj_gemm_kernel<<<gGemm, blk, 0, stream>>>(input2, Wv2, v2t, 1);

  dim3 gAttn(BATCH * HEADS * (SEQ / 128), 1, 1);   // 256 workgroups
  // output1: queries from input1
  flash_attn_kernel<<<gAttn, blk, 0, stream>>>(q1, k1, v1t, k2, v2t,
                                               mask1, mask1, mask2, out);
  // output2: queries from input2
  flash_attn_kernel<<<gAttn, blk, 0, stream>>>(q2, k1, v1t, k2, v2t,
                                               mask2, mask1, mask2,
                                               out + (size_t)BATCH * SEQ * HIDDEN);
}
